// Encoder_78889959293176
// MI455X (gfx1250) — matrix operand order repaired and checked by reading, not compiled, not run on hardware
//
#include <hip/hip_runtime.h>
#include <cstdint>
#include <cstddef>

// ---------------- problem constants ----------------
#define B_   16
#define L_   2048
#define D_   512
#define H_   8
#define E_   64
#define CF_  2048
#define M_   (B_ * L_)          // 32768 rows
#define TOPK 22
#define NEG_SLOPE 0.01f

// ---------------- WMMA types ----------------
typedef __attribute__((ext_vector_type(16))) __bf16 v16bf;
typedef __attribute__((ext_vector_type(2)))  __bf16 v2bf;
typedef __attribute__((ext_vector_type(8)))  float  v8f;

// ---------------- GEMM tiling ----------------
#define BM 128
#define BN 64
#define BK 32
#define ACH 32   // LDS chunk stride in elements: 16 payload + 16 pad = 64B (keeps 32B alignment)
#define NIT_A (BM * BK / 2 / 256)   // 8 staging iterations (paired)
#define NIT_B (BK * BN / 256)       // 8 staging iterations
#define LDSZ_A (BM * 2 * ACH)
#define LDSZ_B (BN * 2 * ACH)

// swizzled-position helpers (fragment-major LDS layout)
__device__ __forceinline__ int a_sw(int r, int c) {   // r=row 0..BM-1, c=k 0..31
  int h   = (c >> 3) & 1;
  int pos = (c & 7) | ((c >> 4) << 3);
  return (r * 2 + h) * ACH + pos;
}
__device__ __forceinline__ int b_sw(int r, int c) {   // r=k 0..31, c=n 0..BN-1
  return (c * 2 + ((r >> 3) & 1)) * ACH + ((r & 7) | ((r >> 4) << 3));
}

// One fragment = one contiguous 32B chunk per lane (2x ds_load_b128)
__device__ __forceinline__ v16bf frag_ld(const __bf16* Ls, int base, int lane) {
  int chunk = (base + (lane & 15)) * 2 + (lane >> 4);
  return *(const v16bf*)(Ls + chunk * ACH);
}

__device__ __forceinline__ void mma_tile(const __bf16* As, const __bf16* Bs,
                                         v8f acc[2][2], int lane, int waveM, int waveN) {
  v16bf a0 = frag_ld(As, waveM * 32,      lane);
  v16bf a1 = frag_ld(As, waveM * 32 + 16, lane);
  v16bf b0 = frag_ld(Bs, waveN * 32,      lane);
  v16bf b1 = frag_ld(Bs, waveN * 32 + 16, lane);
  acc[0][0] = __builtin_amdgcn_wmma_f32_16x16x32_bf16(false, a0, false, b0, (short)0, acc[0][0], false, false);
  acc[0][1] = __builtin_amdgcn_wmma_f32_16x16x32_bf16(false, a0, false, b1, (short)0, acc[0][1], false, false);
  acc[1][0] = __builtin_amdgcn_wmma_f32_16x16x32_bf16(false, a1, false, b0, (short)0, acc[1][0], false, false);
  acc[1][1] = __builtin_amdgcn_wmma_f32_16x16x32_bf16(false, a1, false, b1, (short)0, acc[1][1], false, false);
}

// C layout: lane n=l&15; VGPR v -> row m = v + 8*(l>=16).  Variadic: commas in body survive.
#define EPILOGUE_LOOP(...)                                                    \
  for (int fm = 0; fm < 2; ++fm)                                              \
    for (int fn = 0; fn < 2; ++fn) {                                          \
      int n = nBlock + waveN * 32 + fn * 16 + (lane & 15);                    \
      _Pragma("unroll")                                                       \
      for (int v = 0; v < 8; ++v) {                                           \
        int m = mBlock + waveM * 32 + fm * 16 + v + ((lane >> 4) << 3);       \
        float cval = acc[fm][fn][v];                                          \
        __VA_ARGS__                                                           \
      }                                                                       \
    }

// ============ GEMM 1: projection  Out[(b*512+n)][l] = x @ W + b  (stored [B,H,E,L]) ============
__global__ __launch_bounds__(256) void gemm_proj_kernel(const float* __restrict__ X,
                                                        const float* __restrict__ W,
                                                        const float* __restrict__ bias,
                                                        float* __restrict__ Out) {
  __shared__ alignas(64) __bf16 As[2][LDSZ_A];
  __shared__ alignas(64) __bf16 Bs[2][LDSZ_B];
  int tid = threadIdx.x, lane = tid & 31, wave = tid >> 5;
  int waveM = wave >> 1, waveN = wave & 1;
  int mBlock = blockIdx.x * BM, nBlock = blockIdx.y * BN;
  v8f acc[2][2] = {};

  auto stage = [&](int kt, int p) {
    float2 va[NIT_A];
    float  vb[NIT_B];
#pragma unroll
    for (int it = 0; it < NIT_A; ++it) {                   // batched loads
      int i = tid + it * 256;
      int r = i >> 4, cp = (i & 15) * 2;
      va[it] = *(const float2*)(X + (size_t)(mBlock + r) * D_ + kt + cp);
    }
#pragma unroll
    for (int it = 0; it < NIT_B; ++it) {
      int i = tid + it * 256;
      int r = i >> 6, c = i & 63;
      vb[it] = W[(size_t)(kt + r) * D_ + nBlock + c];
    }
#pragma unroll
    for (int it = 0; it < NIT_A; ++it) {                   // convert + LDS store
      int i = tid + it * 256;
      int r = i >> 4, cp = (i & 15) * 2;
      v2bf pk = { (__bf16)va[it].x, (__bf16)va[it].y };
      *(v2bf*)(&As[p][a_sw(r, cp)]) = pk;
    }
#pragma unroll
    for (int it = 0; it < NIT_B; ++it) {
      int i = tid + it * 256;
      int r = i >> 6, c = i & 63;
      Bs[p][b_sw(r, c)] = (__bf16)vb[it];
    }
  };

  stage(0, 0);
  __syncthreads();
  int p = 0;
  for (int kt = 0; kt < D_; kt += BK) {
    if (kt + BK < D_) stage(kt + BK, p ^ 1);               // prefetch next tile
    mma_tile(As[p], Bs[p], acc, lane, waveM, waveN);
    __syncthreads();
    p ^= 1;
  }
  // transposed store: 8 accumulator lanes are contiguous in l -> vector store
#pragma unroll
  for (int fm = 0; fm < 2; ++fm)
#pragma unroll
    for (int fn = 0; fn < 2; ++fn) {
      int n  = nBlock + waveN * 32 + fn * 16 + (lane & 15);
      int m0 = mBlock + waveM * 32 + fm * 16 + ((lane >> 4) << 3);
      int b = m0 >> 11, l0 = m0 & (L_ - 1);
      v8f r = acc[fm][fn] + bias[n];
      *(v8f*)(&Out[((size_t)(b * D_ + n) << 11) + l0]) = r;
    }
}

// ============ GEMM 2: attention output  Out = agg(T) @ Wo + bo + x  ============
__global__ __launch_bounds__(256) void gemm_attnout_kernel(const float* __restrict__ Agg,
                                                           const float* __restrict__ Wo,
                                                           const float* __restrict__ bo,
                                                           const float* __restrict__ Xres,
                                                           float* __restrict__ Out) {
  __shared__ alignas(64) __bf16 As[2][LDSZ_A];
  __shared__ alignas(64) __bf16 Bs[2][LDSZ_B];
  int tid = threadIdx.x, lane = tid & 31, wave = tid >> 5;
  int waveM = wave >> 1, waveN = wave & 1;
  int mBlock = blockIdx.x * BM, nBlock = blockIdx.y * BN;
  v8f acc[2][2] = {};

  auto stage = [&](int kt, int p) {
    float2 va[NIT_A];
    float  vb[NIT_B];
#pragma unroll
    for (int it = 0; it < NIT_A; ++it) {                   // gather (stride L)
      int i = tid + it * 256;
      int r = i >> 4, cp = (i & 15) * 2;
      int gm = mBlock + r;
      int b = gm >> 11, l = gm & (L_ - 1);
      size_t base = ((size_t)(b * D_ + kt + cp) << 11) + l;
      va[it].x = Agg[base];
      va[it].y = Agg[base + ((size_t)1 << 11)];
    }
#pragma unroll
    for (int it = 0; it < NIT_B; ++it) {
      int i = tid + it * 256;
      int r = i >> 6, c = i & 63;
      vb[it] = Wo[(size_t)(kt + r) * D_ + nBlock + c];
    }
#pragma unroll
    for (int it = 0; it < NIT_A; ++it) {
      int i = tid + it * 256;
      int r = i >> 4, cp = (i & 15) * 2;
      v2bf pk = { (__bf16)va[it].x, (__bf16)va[it].y };
      *(v2bf*)(&As[p][a_sw(r, cp)]) = pk;
    }
#pragma unroll
    for (int it = 0; it < NIT_B; ++it) {
      int i = tid + it * 256;
      int r = i >> 6, c = i & 63;
      Bs[p][b_sw(r, c)] = (__bf16)vb[it];
    }
  };

  stage(0, 0);
  __syncthreads();
  int p = 0;
  for (int kt = 0; kt < D_; kt += BK) {
    if (kt + BK < D_) stage(kt + BK, p ^ 1);
    mma_tile(As[p], Bs[p], acc, lane, waveM, waveN);
    __syncthreads();
    p ^= 1;
  }
  EPILOGUE_LOOP(
    size_t o = (size_t)m * D_ + n;
    Out[o] = cval + bo[n] + Xres[o];
  )
}

// ============ GEMM 3: conv1 (im2col K=1536) -> LeakyReLU -> bf16 h1 ============
__global__ __launch_bounds__(256) void gemm_conv1_kernel(const float* __restrict__ S,
                                                         const float* __restrict__ W1,
                                                         __bf16* __restrict__ H1) {
  __shared__ alignas(64) __bf16 As[2][LDSZ_A];
  __shared__ alignas(64) __bf16 Bs[2][LDSZ_B];
  int tid = threadIdx.x, lane = tid & 31, wave = tid >> 5;
  int waveM = wave >> 1, waveN = wave & 1;
  int mBlock = blockIdx.x * BM, nBlock = blockIdx.y * BN;
  v8f acc[2][2] = {};

  auto stage = [&](int kt, int p) {
    float2 va[NIT_A];
    float  vb[NIT_B];
#pragma unroll
    for (int it = 0; it < NIT_A; ++it) {
      int i = tid + it * 256;
      int r = i >> 4, cp = (i & 15) * 2;
      int gk = kt + cp;                       // even; pair never crosses a k3 boundary
      int k3 = gk >> 9, ch = gk & (D_ - 1);
      int gm = mBlock + r;
      int b = gm >> 11, l = gm & (L_ - 1);
      int ll = l + k3 - 1; ll = ll < 0 ? 0 : (ll > L_ - 1 ? L_ - 1 : ll);
      va[it] = *(const float2*)(S + ((size_t)((b << 11) + ll)) * D_ + ch);
    }
#pragma unroll
    for (int it = 0; it < NIT_B; ++it) {
      int i = tid + it * 256;
      int r = i >> 6, c = i & 63;
      vb[it] = W1[(size_t)(kt + r) * CF_ + nBlock + c];
    }
#pragma unroll
    for (int it = 0; it < NIT_A; ++it) {
      int i = tid + it * 256;
      int r = i >> 4, cp = (i & 15) * 2;
      v2bf pk = { (__bf16)va[it].x, (__bf16)va[it].y };
      *(v2bf*)(&As[p][a_sw(r, cp)]) = pk;
    }
#pragma unroll
    for (int it = 0; it < NIT_B; ++it) {
      int i = tid + it * 256;
      int r = i >> 6, c = i & 63;
      Bs[p][b_sw(r, c)] = (__bf16)vb[it];
    }
  };

  stage(0, 0);
  __syncthreads();
  int p = 0;
  for (int kt = 0; kt < 3 * D_; kt += BK) {
    if (kt + BK < 3 * D_) stage(kt + BK, p ^ 1);
    mma_tile(As[p], Bs[p], acc, lane, waveM, waveN);
    __syncthreads();
    p ^= 1;
  }
  EPILOGUE_LOOP(
    float h = cval > 0.f ? cval : NEG_SLOPE * cval;
    H1[(size_t)m * CF_ + n] = (__bf16)h;
  )
}

// ============ GEMM 4: conv2 (im2col K=6144, bf16 A) -> LeakyReLU -> += season (in-place) ============
__global__ __launch_bounds__(256) void gemm_conv2_kernel(const __bf16* __restrict__ H1,
                                                         const float* __restrict__ W2,
                                                         float* __restrict__ SP) {
  __shared__ alignas(64) __bf16 As[2][LDSZ_A];
  __shared__ alignas(64) __bf16 Bs[2][LDSZ_B];
  int tid = threadIdx.x, lane = tid & 31, wave = tid >> 5;
  int waveM = wave >> 1, waveN = wave & 1;
  int mBlock = blockIdx.x * BM, nBlock = blockIdx.y * BN;
  v8f acc[2][2] = {};

  auto stage = [&](int kt, int p) {
    v2bf  va[NIT_A];
    float vb[NIT_B];
#pragma unroll
    for (int it = 0; it < NIT_A; ++it) {
      int i = tid + it * 256;
      int r = i >> 4, cp = (i & 15) * 2;
      int gk = kt + cp;                       // even; pair never crosses a k3 boundary
      int k3 = gk >> 11, ch = gk & (CF_ - 1);
      int gm = mBlock + r;
      int b = gm >> 11, l = gm & (L_ - 1);
      int ll = l + k3 - 1; ll = ll < 0 ? 0 : (ll > L_ - 1 ? L_ - 1 : ll);
      va[it] = *(const v2bf*)(H1 + ((size_t)((b << 11) + ll)) * CF_ + ch);   // pure copy, no cvt
    }
#pragma unroll
    for (int it = 0; it < NIT_B; ++it) {
      int i = tid + it * 256;
      int r = i >> 6, c = i & 63;
      vb[it] = W2[(size_t)(kt + r) * D_ + nBlock + c];
    }
#pragma unroll
    for (int it = 0; it < NIT_A; ++it) {
      int i = tid + it * 256;
      int r = i >> 4, cp = (i & 15) * 2;
      *(v2bf*)(&As[p][a_sw(r, cp)]) = va[it];
    }
#pragma unroll
    for (int it = 0; it < NIT_B; ++it) {
      int i = tid + it * 256;
      int r = i >> 6, c = i & 63;
      Bs[p][b_sw(r, c)] = (__bf16)vb[it];
    }
  };

  stage(0, 0);
  __syncthreads();
  int p = 0;
  for (int kt = 0; kt < 3 * CF_; kt += BK) {
    if (kt + BK < 3 * CF_) stage(kt + BK, p ^ 1);
    mma_tile(As[p], Bs[p], acc, lane, waveM, waveN);
    __syncthreads();
    p ^= 1;
  }
  EPILOGUE_LOOP(
    float h = cval > 0.f ? cval : NEG_SLOPE * cval;
    size_t o = (size_t)m * D_ + n;
    SP[o] = h + SP[o];                 // pre = leaky(conv2) + season
  )
}

// ---------------- FFT autocorrelation ----------------
struct cpx { float x, y; };

__device__ __forceinline__ void fft_inplace(cpx* Z, int tid, float sign) {
#pragma unroll 1
  for (int sh = 0; sh < 11; ++sh) {
    int half = 1 << sh;
    float ang0 = sign * 6.28318530717958647692f / (float)(half << 1);
#pragma unroll
    for (int r = 0; r < 4; ++r) {
      int i = tid + (r << 8);                     // 0..1023 butterflies
      int blk = i >> sh;
      int pos = i & (half - 1);
      int i0 = (blk << (sh + 1)) + pos;
      int i1 = i0 + half;
      float s, c;
      __sincosf(ang0 * (float)pos, &s, &c);
      cpx a = Z[i0], b = Z[i1];
      cpx t = { c * b.x - s * b.y, c * b.y + s * b.x };
      Z[i0] = { a.x + t.x, a.y + t.y };
      Z[i1] = { a.x - t.x, a.y - t.y };
    }
    __syncthreads();
  }
}

__global__ __launch_bounds__(256) void autocorr_fft_kernel(const float* __restrict__ Q,
                                                           const float* __restrict__ K,
                                                           float* __restrict__ meanv) {
  __shared__ cpx Z[L_];
  int seq = blockIdx.x;                 // (b*H + h)*E + e
  int b = seq >> 9;                     // / (H*E)
  int tid = threadIdx.x;
  const float* q = Q + (size_t)seq * L_;
  const float* k = K + (size_t)seq * L_;
  // load packed z = q + i*k, bit-reversed (11 bits) for DIT
#pragma unroll
  for (int r = 0; r < 8; ++r) {
    int j  = tid + (r << 8);
    int rj = (int)(__brev((unsigned)j) >> 21);
    Z[rj] = { q[j], k[j] };
  }
  __syncthreads();
  fft_inplace(Z, tid, -1.0f);           // forward FFT
  // cross spectrum C = Qf * conj(Kf) from packed transform
  cpx Cv[8];
#pragma unroll
  for (int r = 0; r < 8; ++r) {
    int j = tid + (r << 8);
    cpx zj = Z[j];
    cpx zr = Z[(L_ - j) & (L_ - 1)];
    cpx Qf = { 0.5f * (zj.x + zr.x), 0.5f * (zj.y - zr.y) };
    cpx df = { zj.x - zr.x, zj.y + zr.y };
    cpx Kf = { 0.5f * df.y, -0.5f * df.x };         // (-0.5i)*df
    Cv[r].x = Qf.x * Kf.x + Qf.y * Kf.y;            // Qf * conj(Kf)
    Cv[r].y = Qf.y * Kf.x - Qf.x * Kf.y;
  }
  __syncthreads();
#pragma unroll
  for (int r = 0; r < 8; ++r) {
    int j  = tid + (r << 8);
    int rj = (int)(__brev((unsigned)j) >> 21);
    Z[rj] = Cv[r];
  }
  __syncthreads();
  fft_inplace(Z, tid, +1.0f);           // inverse FFT (unnormalized)
  const float scale = 1.0f / ((float)L_ * (float)(H_ * E_));
#pragma unroll
  for (int r = 0; r < 8; ++r) {
    int j = tid + (r << 8);
    atomicAdd(meanv + b * L_ + j, Z[j].x * scale);
  }
}

// ---------------- top-k + softmax (single block) ----------------
__global__ __launch_bounds__(256) void topk_kernel(const float* __restrict__ meanv,
                                                   int* __restrict__ idx_out,
                                                   float* __restrict__ sw_out) {
  __shared__ float mb[L_];
  __shared__ float rv[256];
  __shared__ int   ri[256];
  __shared__ int   sidx[TOPK];
  int tid = threadIdx.x;
  for (int r = 0; r < 8; ++r) {
    int l = tid + (r << 8);
    float s = 0.f;
    for (int b = 0; b < B_; ++b) s += meanv[b * L_ + l];
    mb[l] = s * (1.0f / (float)B_);
  }
  __syncthreads();
  for (int it = 0; it < TOPK; ++it) {
    float best = -1e30f; int bi = 0;
    for (int r = 0; r < 8; ++r) {
      int l = tid + (r << 8);
      if (mb[l] > best) { best = mb[l]; bi = l; }
    }
    rv[tid] = best; ri[tid] = bi;
    __syncthreads();
    for (int s = 128; s > 0; s >>= 1) {
      if (tid < s && rv[tid + s] > rv[tid]) { rv[tid] = rv[tid + s]; ri[tid] = ri[tid + s]; }
      __syncthreads();
    }
    if (tid == 0) { sidx[it] = ri[0]; mb[ri[0]] = -1e30f; }
    __syncthreads();
  }
  if (tid < TOPK) idx_out[tid] = sidx[tid];
  if (tid < B_) {
    int b = tid;
    float w[TOPK], mx = -1e30f;
    for (int i = 0; i < TOPK; ++i) { w[i] = meanv[b * L_ + sidx[i]]; mx = fmaxf(mx, w[i]); }
    float sum = 0.f;
    for (int i = 0; i < TOPK; ++i) { w[i] = __expf(w[i] - mx); sum += w[i]; }
    float inv = 1.0f / sum;
    for (int i = 0; i < TOPK; ++i) sw_out[b * TOPK + i] = w[i] * inv;
  }
}

// ---------------- delay-roll aggregation ----------------
__global__ __launch_bounds__(256) void agg_kernel(const float* __restrict__ V,
                                                  const int* __restrict__ idx,
                                                  const float* __restrict__ sw,
                                                  float* __restrict__ Agg) {
  __shared__ int   di[TOPK];
  __shared__ float dw[TOPK];
  int seq = blockIdx.x;
  int b = seq >> 9;
  int tid = threadIdx.x;
  if (tid < TOPK) { di[tid] = idx[tid]; dw[tid] = sw[b * TOPK + tid]; }
  __syncthreads();
  const float* v = V + (size_t)seq * L_;
  float* o = Agg + (size_t)seq * L_;
#pragma unroll 1
  for (int r = 0; r < 8; ++r) {
    int l = tid + (r << 8);
    float a = 0.f;
#pragma unroll 1
    for (int i = 0; i < TOPK; ++i) {
      int s = l + di[i]; if (s >= L_) s -= L_;
      a += dw[i] * v[s];
    }
    o[l] = a;
  }
}

// ---------------- series decomposition (25-tap replicate-padded moving mean) ----------------
__global__ __launch_bounds__(256) void decompose_kernel(const float* __restrict__ X,
                                                        float* __restrict__ S) {
  int gid = blockIdx.x * blockDim.x + threadIdx.x;   // 0..B*D-1
  int b = gid >> 9, d = gid & (D_ - 1);
  const float* x = X + (size_t)b * L_ * D_ + d;
  float* s = S + (size_t)b * L_ * D_ + d;
  float sum = 13.0f * x[0];
  for (int t = 1; t <= 12; ++t) sum += x[t * D_];
  for (int l = 0; l < L_; ++l) {
    float xc = x[(size_t)l * D_];
    s[(size_t)l * D_] = xc - sum * (1.0f / 25.0f);
    int ip = l + 13; if (ip > L_ - 1) ip = L_ - 1;
    int im = l - 12; if (im < 0) im = 0;
    sum += x[(size_t)ip * D_] - x[(size_t)im * D_];
  }
}

// ---------------- LayerNorm (one wave32 per row of 512) ----------------
__global__ __launch_bounds__(256) void ln_kernel(const float* __restrict__ X,
                                                 const float* __restrict__ g,
                                                 const float* __restrict__ bta,
                                                 float* __restrict__ out) {
  int row  = (int)((blockIdx.x * blockDim.x + threadIdx.x) >> 5);
  int lane = threadIdx.x & 31;
  if (row >= M_) return;
  const float* x = X + (size_t)row * D_;
  float vals[16], sum = 0.f, sq = 0.f;
#pragma unroll
  for (int i = 0; i < 16; ++i) {
    float v = x[lane + i * 32];
    vals[i] = v; sum += v; sq += v * v;
  }
#pragma unroll
  for (int o = 16; o > 0; o >>= 1) {
    sum += __shfl_xor(sum, o, 32);
    sq  += __shfl_xor(sq,  o, 32);
  }
  float mu  = sum * (1.0f / (float)D_);
  float var = sq * (1.0f / (float)D_) - mu * mu;
  float inv = rsqrtf(var + 1e-5f);
  float* o_ = out + (size_t)row * D_;
#pragma unroll
  for (int i = 0; i < 16; ++i) {
    int d = lane + i * 32;
    o_[d] = (vals[i] - mu) * inv * g[d] + bta[d];
  }
}

__global__ void zero_kernel(float* p, int n) {
  int i = blockIdx.x * blockDim.x + threadIdx.x;
  if (i < n) p[i] = 0.f;
}

// ---------------- launch ----------------
extern "C" void kernel_launch(void* const* d_in, const int* in_sizes, int n_in,
                              void* d_out, int out_size, void* d_ws, size_t ws_size,
                              hipStream_t stream) {
  (void)in_sizes; (void)n_in; (void)out_size; (void)ws_size;
  const float* x   = (const float*)d_in[0];
  const float* Wq  = (const float*)d_in[1];
  const float* bq  = (const float*)d_in[2];
  const float* Wk  = (const float*)d_in[3];
  const float* bk  = (const float*)d_in[4];
  const float* Wv  = (const float*)d_in[5];
  const float* bv  = (const float*)d_in[6];
  const float* Wo  = (const float*)d_in[7];
  const float* bo  = (const float*)d_in[8];
  const float* w1  = (const float*)d_in[9];
  const float* w2  = (const float*)d_in[10];
  const float* lng = (const float*)d_in[11];
  const float* lnb = (const float*)d_in[12];
  float* out = (float*)d_out;

  char* ws = (char*)d_ws;
  const size_t SEQB = (size_t)M_ * D_ * sizeof(float);     // 64 MiB
  float* bufQ  = (float*)(ws);                             // q -> agg -> h1(lo) -> season2
  float* bufK  = (float*)(ws + SEQB);                      // k -> autocorr_x -> h1(hi)
  float* bufV  = (float*)(ws + 2 * SEQB);                  // v -> season -> pre
  float* meanv = (float*)(ws + 3 * SEQB);                  // [B,L]
  int*   idxb  = (int*)  (ws + 3 * SEQB + 131072);         // [TOPK]
  float* swb   = (float*)(ws + 3 * SEQB + 131072 + 256);   // [B,TOPK]
  __bf16* h1 = (__bf16*)bufQ;                              // bf16 [M, CF] spans bufQ+bufK

  dim3 blk(256);
  dim3 gP(M_ / BM, D_ / BN);                               // 256 x 8
  dim3 gC1(M_ / BM, CF_ / BN);                             // 256 x 32

  zero_kernel<<<dim3(128), blk, 0, stream>>>(meanv, B_ * L_);
  gemm_proj_kernel<<<gP, blk, 0, stream>>>(x, Wq, bq, bufQ);
  gemm_proj_kernel<<<gP, blk, 0, stream>>>(x, Wk, bk, bufK);
  gemm_proj_kernel<<<gP, blk, 0, stream>>>(x, Wv, bv, bufV);
  autocorr_fft_kernel<<<dim3(B_ * H_ * E_), blk, 0, stream>>>(bufQ, bufK, meanv);
  topk_kernel<<<dim3(1), blk, 0, stream>>>(meanv, idxb, swb);
  agg_kernel<<<dim3(B_ * H_ * E_), blk, 0, stream>>>(bufV, idxb, swb, bufQ);
  gemm_attnout_kernel<<<gP, blk, 0, stream>>>(bufQ, Wo, bo, x, bufK);       // autocorr_x
  decompose_kernel<<<dim3(32), blk, 0, stream>>>(bufK, bufV);               // season_first
  gemm_conv1_kernel<<<gC1, blk, 0, stream>>>(bufV, w1, h1);                 // h1 bf16
  gemm_conv2_kernel<<<gP, blk, 0, stream>>>(h1, w2, bufV);                  // pre (in place)
  decompose_kernel<<<dim3(32), blk, 0, stream>>>(bufV, bufQ);               // season_2
  ln_kernel<<<dim3(M_ * 32 / 256), blk, 0, stream>>>(bufQ, lng, lnb, out);
}
